// Attention_29566554866217
// MI455X (gfx1250) — matrix operand order repaired and checked by reading, not compiled, not run on hardware
//
#include <hip/hip_runtime.h>
#include <stdint.h>

// ---------------------------------------------------------------------------
// Problem constants (match reference):  B=8, S_ENC=S_DEC=2048, D_ENC=D_DEC=1024
// ---------------------------------------------------------------------------
#define NB   8
#define SEQ  2048
#define DIM  1024

typedef __attribute__((ext_vector_type(16))) __bf16 v16bf;
typedef __attribute__((ext_vector_type(8)))  float  v8f;

// f32 -> bf16 with round-to-nearest-even (bit twiddle; deterministic)
static __device__ __forceinline__ uint16_t f2bf(float f) {
    uint32_t u = __float_as_uint(f);
    uint32_t r = u + 0x7FFFu + ((u >> 16) & 1u);
    return (uint16_t)(r >> 16);
}
static __device__ __forceinline__ uint32_t pack2(float lo, float hi) {
    return (uint32_t)f2bf(lo) | ((uint32_t)f2bf(hi) << 16);
}

// 8 dwords <-> one wave32 WMMA 16-bit A/B fragment (16 bf16 per lane)
union Frag16 {
    uint32_t u[8];
    uint4    q[2];
    v16bf    v;
};

static __device__ __forceinline__ v8f wmma_bf16(v16bf a, v16bf b, v8f c) {
    // 8 args: (neg_a, A, neg_b, B, c_mod, C, reuse_a, reuse_b)
    return __builtin_amdgcn_wmma_f32_16x16x32_bf16(
        false, a, false, b, (short)0, c, false, false);
}

// ---------------------------------------------------------------------------
// Phase 1: projection GEMM.  out[s][d] = sum_e X[s][e] * W[d][e]
//   MODE 0 (enc): store octet-packed transposed  encO[b][d/8][s] (uint4 = 8 bf16,
//                 d fastest inside the uint4) -> phase-2 B fragment = 2 B128 loads
//   MODE 1 (dec): store row-major bf16           dec[b][s][d]
// One wave per 16x16 output tile; K loop of 32 with v_wmma_f32_16x16x32_bf16.
// ---------------------------------------------------------------------------
template <int MODE>
__global__ __launch_bounds__(256) void proj_kernel(const float* __restrict__ X,
                                                   const float* __restrict__ W,
                                                   void* __restrict__ out)
{
    const int lane = threadIdx.x & 31;
    const int wave = threadIdx.x >> 5;
    const int g    = lane >> 4;     // half-wave group (K-range selector)
    const int ln   = lane & 15;     // row/col within 16

    const int TD   = DIM / 16;      // 64 tiles across output-dim
    const int TS   = SEQ / 16;      // 128 tiles across sequence

    int tile = blockIdx.x * 8 + wave;
    int dT   = tile % TD;
    int rest = tile / TD;
    int sT   = rest % TS;
    int b    = rest / TS;

    const float* xrow = X + ((size_t)b * SEQ + (size_t)(sT * 16 + ln)) * DIM;
    const float* wrow = W + (size_t)(dT * 16 + ln) * DIM;

    v8f acc = {};

    for (int kk = 0; kk < DIM; kk += 32) {
        // ---- A fragment: row m = ln of X; K elems [kk+g*8 .. +7] and [kk+16+g*8 .. +7]
        float4 a0 = *(const float4*)(xrow + kk + g * 8);
        float4 a1 = *(const float4*)(xrow + kk + g * 8 + 4);
        float4 a2 = *(const float4*)(xrow + kk + 16 + g * 8);
        float4 a3 = *(const float4*)(xrow + kk + 16 + g * 8 + 4);
        Frag16 A;
        A.u[0] = pack2(a0.x, a0.y);  A.u[1] = pack2(a0.z, a0.w);
        A.u[2] = pack2(a1.x, a1.y);  A.u[3] = pack2(a1.z, a1.w);
        A.u[4] = pack2(a2.x, a2.y);  A.u[5] = pack2(a2.z, a2.w);
        A.u[6] = pack2(a3.x, a3.y);  A.u[7] = pack2(a3.z, a3.w);

        // ---- B fragment: col n = ln from W; K elems [kk+g*8 .. +7] and [kk+16+g*8 .. +7]
        float4 b0 = *(const float4*)(wrow + kk + g * 8);
        float4 b1 = *(const float4*)(wrow + kk + g * 8 + 4);
        float4 b2 = *(const float4*)(wrow + kk + 16 + g * 8);
        float4 b3 = *(const float4*)(wrow + kk + 16 + g * 8 + 4);
        Frag16 Bf;
        Bf.u[0] = pack2(b0.x, b0.y);  Bf.u[1] = pack2(b0.z, b0.w);
        Bf.u[2] = pack2(b1.x, b1.y);  Bf.u[3] = pack2(b1.z, b1.w);
        Bf.u[4] = pack2(b2.x, b2.y);  Bf.u[5] = pack2(b2.z, b2.w);
        Bf.u[6] = pack2(b3.x, b3.y);  Bf.u[7] = pack2(b3.z, b3.w);

        acc = wmma_bf16(A.v, Bf.v, acc);
    }

    // C/D layout: VGPR j holds row M = g*8+j, col N = ln.
    if (MODE == 0) {
        // Assemble 8 consecutive d per (s) from 8 adjacent lanes, then lanes
        // ln%8==0 write 8 contiguous uint4 (128 B) into encO[b][d/8][s].
        uint32_t pk[8], t2[8], t4[8], t6[8];
        #pragma unroll
        for (int j = 0; j < 8; ++j) {
            float v0 = acc[j];
            float v1 = __shfl_xor(v0, 1, 32);   // partner column (d xor 1)
            pk[j] = pack2(v0, v1);              // even lanes: (d, d+1)
        }
        #pragma unroll
        for (int j = 0; j < 8; ++j) {
            t2[j] = __shfl_xor(pk[j], 2, 32);   // (d+2, d+3)
            t4[j] = __shfl_xor(pk[j], 4, 32);   // (d+4, d+5)
            t6[j] = __shfl_xor(pk[j], 6, 32);   // (d+6, d+7)
        }
        if ((ln & 7) == 0) {
            int doct = (dT * 16 + ln) >> 3;     // d/8
            int s0   = sT * 16 + g * 8;
            uint4* dst = (uint4*)out + ((size_t)b * (DIM / 8) + doct) * SEQ + s0;
            #pragma unroll
            for (int j = 0; j < 8; ++j)
                dst[j] = make_uint4(pk[j], t2[j], t4[j], t6[j]);
        }
    } else {
        // row-major bf16 dec[b][t][d]
        uint16_t* dst = (uint16_t*)out;
        int d = dT * 16 + ln;
        #pragma unroll
        for (int j = 0; j < 8; ++j) {
            int t = sT * 16 + g * 8 + j;
            dst[((size_t)b * SEQ + t) * DIM + d] = f2bf(acc[j]);
        }
    }
}

// ---------------------------------------------------------------------------
// Phase 2: attn[t][s] = sum_d dec[t][d]*enc[s][d], masked softmax over s.
// Block = 16 decoder rows x 2048 encoder cols. 8 waves x 16 column tiles each,
// accumulators in registers (16 x v8f). Operands are bf16 in L2 (64 MB total).
// B fragment = 2 x B128 loads thanks to octet-packed encO layout.
// Mask semantics match reference: logit = attn*mask (+const EPSILON cancels).
// ---------------------------------------------------------------------------
__global__ __launch_bounds__(256) void attn_softmax_kernel(
    const uint4*    __restrict__ encO,   // [NB][DIM/8][SEQ]  octet-packed bf16
    const uint16_t* __restrict__ decq,   // [NB][SEQ][DIM]    bf16
    const int*      __restrict__ h_len,
    const int*      __restrict__ y_len,
    float*          __restrict__ out)    // [NB][SEQ][SEQ]
{
    __shared__ float red[8][16];

    const int lane = threadIdx.x & 31;
    const int wave = threadIdx.x >> 5;
    const int g    = lane >> 4;
    const int ln   = lane & 15;

    const int b  = blockIdx.x / (SEQ / 16);
    const int sT = blockIdx.x % (SEQ / 16);

    const uint16_t* arow = decq + ((size_t)b * SEQ + (size_t)(sT * 16 + ln)) * DIM;
    const uint4*    encB = encO + (size_t)b * (DIM / 8) * SEQ;

    v8f acc[16];
    #pragma unroll
    for (int i = 0; i < 16; ++i) acc[i] = (v8f){};

    for (int kk = 0; kk < DIM; kk += 32) {
        __builtin_prefetch(arow + kk + 32, 0, 1);   // global_prefetch_b8 next A slice

        // A fragment (dec row m=ln): two B128 loads of contiguous bf16
        Frag16 A;
        A.q[0] = *(const uint4*)(arow + kk + g * 16);
        A.q[1] = *(const uint4*)(arow + kk + g * 16 + 8);

        // B fragment base: octet index do = (kk + g*16)/8 = kk/8 + 2g
        const uint4* ebase = encB + (size_t)(kk / 8 + g * 2) * SEQ;

        #pragma unroll
        for (int i = 0; i < 16; ++i) {
            int s = (wave + i * 8) * 16 + ln;       // B col n = ln of tile cT
            Frag16 Bf;
            Bf.q[0] = ebase[s];                     // d = base .. base+7
            Bf.q[1] = ebase[SEQ + s];               // d = base+8 .. base+15
            acc[i] = wmma_bf16(A.v, Bf.v, acc[i]);
        }
    }

    const int hl = h_len[b];
    const int yl = y_len[b];

    // apply mask: invalid (t,s) -> logit 0 (reference multiplies scores by mask)
    #pragma unroll
    for (int i = 0; i < 16; ++i) {
        int s = (wave + i * 8) * 16 + ln;
        bool sv = s < hl;
        #pragma unroll
        for (int j = 0; j < 8; ++j) {
            int t = sT * 16 + g * 8 + j;
            if (!(sv && (t < yl))) acc[i][j] = 0.0f;
        }
    }

    // ---- row max: per-lane over 16 tiles, shuffle over 16-lane group, LDS over waves
    float mx[8];
    #pragma unroll
    for (int j = 0; j < 8; ++j) {
        float m = acc[0][j];
        #pragma unroll
        for (int i = 1; i < 16; ++i) m = fmaxf(m, acc[i][j]);
        for (int d = 1; d < 16; d <<= 1) m = fmaxf(m, __shfl_xor(m, d, 32));
        mx[j] = m;
    }
    if (ln == 0) {
        #pragma unroll
        for (int j = 0; j < 8; ++j) red[wave][g * 8 + j] = mx[j];
    }
    __syncthreads();
    #pragma unroll
    for (int j = 0; j < 8; ++j) {
        float m = red[0][g * 8 + j];
        #pragma unroll
        for (int w = 1; w < 8; ++w) m = fmaxf(m, red[w][g * 8 + j]);
        mx[j] = m;
    }
    __syncthreads();   // red reused below

    // ---- exp + row sum
    float sm[8];
    #pragma unroll
    for (int j = 0; j < 8; ++j) {
        float s = 0.0f;
        #pragma unroll
        for (int i = 0; i < 16; ++i) {
            float e = __expf(acc[i][j] - mx[j]);
            acc[i][j] = e;
            s += e;
        }
        for (int d = 1; d < 16; d <<= 1) s += __shfl_xor(s, d, 32);
        sm[j] = s;
    }
    if (ln == 0) {
        #pragma unroll
        for (int j = 0; j < 8; ++j) red[wave][g * 8 + j] = sm[j];
    }
    __syncthreads();
    #pragma unroll
    for (int j = 0; j < 8; ++j) {
        float s = 0.0f;
        #pragma unroll
        for (int w = 0; w < 8; ++w) s += red[w][g * 8 + j];
        sm[j] = 1.0f / s;
    }

    // ---- normalize + store: row t = sT*16+g*8+j, col s; 16 lanes contiguous in s
    #pragma unroll
    for (int i = 0; i < 16; ++i) {
        int s = (wave + i * 8) * 16 + ln;
        #pragma unroll
        for (int j = 0; j < 8; ++j) {
            int t = sT * 16 + g * 8 + j;
            out[((size_t)b * SEQ + t) * SEQ + s] = acc[i][j] * sm[j];
        }
    }
}

// ---------------------------------------------------------------------------
// Host-side launcher
// ---------------------------------------------------------------------------
extern "C" void kernel_launch(void* const* d_in, const int* in_sizes, int n_in,
                              void* d_out, int out_size, void* d_ws, size_t ws_size,
                              hipStream_t stream)
{
    const float* h     = (const float*)d_in[0];   // [8][2048][1024]
    const float* y     = (const float*)d_in[1];   // [8][2048][1024]
    const float* W_enc = (const float*)d_in[2];   // [1024][1024]
    const float* W_dec = (const float*)d_in[3];   // [1024][1024]
    const int*   hlen  = (const int*)d_in[4];     // [8]
    const int*   ylen  = (const int*)d_in[5];     // [8]

    // workspace: encO (32 MB, uint4 octets) then dec bf16 (32 MB)
    uint4*    encO = (uint4*)d_ws;
    uint16_t* decq = (uint16_t*)((char*)d_ws + (size_t)NB * (DIM / 8) * SEQ * sizeof(uint4));

    const int tiles = NB * (SEQ / 16) * (DIM / 16);   // 65536 wave-tiles
    proj_kernel<0><<<tiles / 8, 256, 0, stream>>>(h, W_enc, (void*)encO);
    proj_kernel<1><<<tiles / 8, 256, 0, stream>>>(y, W_dec, (void*)decq);

    attn_softmax_kernel<<<NB * (SEQ / 16), 256, 0, stream>>>(
        encO, decq, hlen, ylen, (float*)d_out);
}
